// MoEPrefixAttention_7541962572299
// MI455X (gfx1250) — compile-verified, matrix operand order checked by reading, not run on hardware
//
#include <hip/hip_runtime.h>
#include <math.h>

// ---------------- CDNA5 (gfx1250) wave32 WMMA bf16 implementation ----------------
typedef __bf16 v16bf __attribute__((ext_vector_type(16)));
typedef __bf16 v8bf  __attribute__((ext_vector_type(8)));
typedef float  v8f   __attribute__((ext_vector_type(8)));

constexpr int Bc = 16, Nc = 773, Cc = 768, Hc = 12, Dc = 64, Pc = 64, Tc = 196;
constexpr int Mr   = Nc + 2 * Pc;   // 901 concatenated KV length
constexpr int Mpad = 928;           // 29 * 32
constexpr int Npad = 832;           // 13 * 64
constexpr int ROWS = Bc * Nc;       // 12368
constexpr float QSCALE = 0.125f;    // D^-0.5

#define DEVINL __device__ __forceinline__

DEVINL v8f zero8() {
  v8f v;
#pragma unroll
  for (int i = 0; i < 8; ++i) v[i] = 0.f;
  return v;
}

DEVINL v8f wmma_bf16(v16bf a, v16bf b, v8f c) {
  // D = A(16x32 bf16) x B(32x16 bf16) + C(16x16 f32)
  return __builtin_amdgcn_wmma_f32_16x16x32_bf16(false, a, false, b, (short)0, c,
                                                 false, false);
}

// A/B fragment load from a row-major [row][k] bf16 matrix (ld = k-stride in elems).
// ISA 16-bit 16x32 layout: lane(l&15) = row; lane(l>>4) selects the two contiguous
// 8-element K chunks {hi*8 .. hi*8+7} and {16+hi*8 .. 16+hi*8+7}  -> two b128 loads.
DEVINL v16bf load_frag(const __bf16* base, int row0, int ld, int k0, int lane) {
  int sub = lane & 15, hi = lane >> 4;
  const __bf16* p = base + (size_t)(row0 + sub) * ld + k0 + hi * 8;
  union { v16bf v; v8bf h[2]; } u;
  u.h[0] = *reinterpret_cast<const v8bf*>(p);
  u.h[1] = *reinterpret_cast<const v8bf*>(p + 16);
  return u.v;
}

DEVINL v16bf load_frag_clamp(const __bf16* base, int row0, int maxrow, int ld,
                             int k0, int lane) {
  int sub = lane & 15, hi = lane >> 4;
  int r = row0 + sub; if (r >= maxrow) r = maxrow - 1;
  const __bf16* p = base + (size_t)r * ld + k0 + hi * 8;
  union { v16bf v; v8bf h[2]; } u;
  u.h[0] = *reinterpret_cast<const v8bf*>(p);
  u.h[1] = *reinterpret_cast<const v8bf*>(p + 16);
  return u.v;
}

// ---------------- helper kernels ----------------
__global__ void cvt_bf16(const float* __restrict__ src, __bf16* __restrict__ dst, int n) {
  int i = blockIdx.x * 256 + threadIdx.x;
  if (i < n) dst[i] = (__bf16)src[i];
}

// h = tanh(emb @ W1^T + b1), emb in {text, image}; hbuf: [2][P][C] fp32
__global__ void prompt_h(const float* __restrict__ text_emb,
                         const float* __restrict__ image_emb,
                         const float* __restrict__ w1, const float* __restrict__ b1,
                         float* __restrict__ hbuf) {
  int idx = blockIdx.x * 256 + threadIdx.x;
  if (idx >= 2 * Pc * Cc) return;
  int e = idx / (Pc * Cc);
  int rem = idx % (Pc * Cc);
  int p = rem / Cc, c = rem % Cc;
  const float* emb = (e == 0 ? text_emb : image_emb) + (size_t)p * Cc;
  const float* w = w1 + (size_t)c * Cc;
  float s = b1[c];
  for (int k = 0; k < Cc; ++k) s += emb[k] * w[k];
  hbuf[idx] = tanhf(s);
}

// p2 = h @ W2^T + b2 ; scatter kp into kbuf rows, vp into vT cols (all batches)
__global__ void prompt_p2(const float* __restrict__ hbuf, const float* __restrict__ w2,
                          const float* __restrict__ b2, __bf16* __restrict__ kbuf,
                          __bf16* __restrict__ vT) {
  int idx = blockIdx.x * 256 + threadIdx.x;
  if (idx >= 2 * Pc * 2 * Cc) return;
  int e = idx / (Pc * 2 * Cc);
  int rem = idx % (Pc * 2 * Cc);
  int p = rem / (2 * Cc), c = rem % (2 * Cc);
  const float* hh = hbuf + ((size_t)e * Pc + p) * Cc;
  const float* w = w2 + (size_t)c * Cc;
  float s = b2[c];
  for (int k = 0; k < Cc; ++k) s += hh[k] * w[k];
  int row = (e == 0) ? p : (Pc + Tc + p);   // text rows [0,P), image rows [P+T,2P+T)
  __bf16 sv = (__bf16)s;
  if (c < Cc) {
    int head = c >> 6, d = c & 63;
    for (int b = 0; b < Bc; ++b)
      kbuf[(((size_t)(b * Hc + head)) * Mpad + row) * Dc + d] = sv;
  } else {
    int cc = c - Cc;
    int head = cc >> 6, d = cc & 63;
    for (int b = 0; b < Bc; ++b)
      vT[(((size_t)(b * Hc + head)) * Dc + d) * Mpad + row] = sv;
  }
}

// ---------------- QKV projection GEMM: [12368 x 2304 x 768] ----------------
// 256 threads = 8 waves (2 row-waves x 4 col-waves); wave tile 32x64:
// per 32-K chunk: 2 A frags + 4 B frags (12 b128) feed 8 WMMAs -> 1.5 loads/WMMA.
__global__ __launch_bounds__(256)
void qkv_gemm(const __bf16* __restrict__ xbf, const __bf16* __restrict__ wq,
              const float* __restrict__ q_bias, const float* __restrict__ v_bias,
              __bf16* __restrict__ qbuf, __bf16* __restrict__ kbuf,
              __bf16* __restrict__ vT) {
  int lane = threadIdx.x & 31, wave = threadIdx.x >> 5;
  int wr = wave >> 2, wc = wave & 3;
  int row0 = blockIdx.x * 64 + wr * 32;
  int col0 = blockIdx.y * 256 + wc * 64;
  v8f acc[2][4];
#pragma unroll
  for (int i = 0; i < 2; ++i)
#pragma unroll
    for (int j = 0; j < 4; ++j) acc[i][j] = zero8();

  for (int kc = 0; kc < Cc; kc += 32) {
    v16bf a0 = load_frag_clamp(xbf, row0,      ROWS, Cc, kc, lane);
    v16bf a1 = load_frag_clamp(xbf, row0 + 16, ROWS, Cc, kc, lane);
#pragma unroll
    for (int j = 0; j < 4; ++j) {
      v16bf bj = load_frag(wq, col0 + j * 16, Cc, kc, lane);
      acc[0][j] = wmma_bf16(a0, bj, acc[0][j]);
      acc[1][j] = wmma_bf16(a1, bj, acc[1][j]);
    }
  }

  int sub = lane & 15, half = lane >> 4;
#pragma unroll
  for (int i = 0; i < 2; ++i)
#pragma unroll
    for (int j = 0; j < 4; ++j)
#pragma unroll
      for (int r = 0; r < 8; ++r) {
        int gr = row0 + i * 16 + r + half * 8;
        if (gr >= ROWS) continue;
        int c = col0 + j * 16 + sub;
        float v = acc[i][j][r];
        int b = gr / Nc, n = gr % Nc;
        if (c < Cc) {                               // ---- Q (scaled)
          int hh = c >> 6, d = c & 63;
          v = (v + q_bias[c]) * QSCALE;
          qbuf[(((size_t)(b * Hc + hh)) * Npad + n) * Dc + d] = (__bf16)v;
        } else if (c < 2 * Cc) {                    // ---- K (row remap)
          int cc = c - Cc;
          int hh = cc >> 6, d = cc & 63;
          int row = (n < Tc) ? (Pc + n) : (2 * Pc + Tc + (n - Tc));
          kbuf[(((size_t)(b * Hc + hh)) * Mpad + row) * Dc + d] = (__bf16)v;
        } else {                                    // ---- V (transposed, row remap)
          int cc = c - 2 * Cc;
          int hh = cc >> 6, d = cc & 63;
          v += v_bias[cc];
          int row = (n < Tc) ? (Pc + n) : (2 * Pc + Tc + (n - Tc));
          vT[(((size_t)(b * Hc + hh)) * Dc + d) * Mpad + row] = (__bf16)v;
        }
      }
}

// ---------------- Flash attention: per (b,h), 16 q-rows per wave ----------------
__global__ __launch_bounds__(128)
void attn_kernel(const __bf16* __restrict__ qbuf, const __bf16* __restrict__ kbuf,
                 const __bf16* __restrict__ vTb, const float* __restrict__ rpb,
                 const float* __restrict__ mask, __bf16* __restrict__ obuf) {
  __shared__ __bf16 ldsP[4][16][32];   // per-wave 16x32 probability tile
  int lane = threadIdx.x & 31, wave = threadIdx.x >> 5;
  int sub = lane & 15, half = lane >> 4;
  int bh = blockIdx.x;
  int b = bh / Hc, h = bh % Hc;
  int q0 = blockIdx.y * 64 + wave * 16;

  const __bf16* qb = qbuf + (size_t)bh * Npad * Dc;
  const __bf16* kb = kbuf + (size_t)bh * Mpad * Dc;
  const __bf16* vb = vTb + (size_t)bh * Dc * Mpad;

  v16bf qa0 = load_frag(qb, q0, Dc, 0, lane);
  v16bf qa1 = load_frag(qb, q0, Dc, 32, lane);

  v8f acc[4];
#pragma unroll
  for (int f = 0; f < 4; ++f) acc[f] = zero8();
  float rmax[8], rsum[8];
#pragma unroll
  for (int r = 0; r < 8; ++r) { rmax[r] = -__builtin_inff(); rsum[r] = 0.f; }

  for (int jt = 0; jt < Mpad / 32; ++jt) {
    int j0 = jt * 32;
    if (j0 + 32 < Mpad)   // CDNA5 prefetch of next K tile (global_prefetch_b8)
      __builtin_prefetch(kb + (size_t)(j0 + 32 + sub) * Dc + half * 32, 0, 1);

    v16bf k00 = load_frag(kb, j0,      Dc, 0,  lane);
    v16bf k01 = load_frag(kb, j0,      Dc, 32, lane);
    v16bf k10 = load_frag(kb, j0 + 16, Dc, 0,  lane);
    v16bf k11 = load_frag(kb, j0 + 16, Dc, 32, lane);
    v8f s0 = wmma_bf16(qa0, k00, zero8()); s0 = wmma_bf16(qa1, k01, s0);
    v8f s1 = wmma_bf16(qa0, k10, zero8()); s1 = wmma_bf16(qa1, k11, s1);

    // relative-position bias + prefix column remap + mask (-inf)
#pragma unroll
    for (int t = 0; t < 2; ++t) {
      int j = j0 + t * 16 + sub;
      int jj; bool valid = true;
      if (j < Pc)                 jj = -1;            // text prompt: bias 0, valid
      else if (j < Pc + Tc)       jj = j - Pc;        // text tokens
      else if (j < 2 * Pc + Tc)   jj = -1;            // image prompt
      else if (j < Mr)            jj = j - 2 * Pc;    // image tokens
      else { jj = -1; valid = false; }                // pad columns
      if (valid && jj >= 0 && mask[b * Nc + jj] == 0.f) valid = false;
      v8f& s = t ? s1 : s0;
#pragma unroll
      for (int r = 0; r < 8; ++r) {
        int qrow = q0 + r + half * 8;
        float bias = 0.f;
        if (jj >= 0 && qrow < Nc)
          bias = rpb[((size_t)h * Nc + qrow) * Nc + jj];
        s[r] = valid ? (s[r] + bias) : -__builtin_inff();
      }
    }

    // ---- online softmax (row = (r, half); columns live across 16 lanes) ----
    float tmax[8];
#pragma unroll
    for (int r = 0; r < 8; ++r) tmax[r] = fmaxf(s0[r], s1[r]);
#pragma unroll
    for (int off = 1; off < 16; off <<= 1)
#pragma unroll
      for (int r = 0; r < 8; ++r)
        tmax[r] = fmaxf(tmax[r], __shfl_xor(tmax[r], off, 32));

    float psum[8];
#pragma unroll
    for (int r = 0; r < 8; ++r) {
      float nm = fmaxf(rmax[r], tmax[r]);
      float alpha = __expf(rmax[r] - nm);
      rmax[r] = nm;
      rsum[r] *= alpha;
      acc[0][r] *= alpha; acc[1][r] *= alpha; acc[2][r] *= alpha; acc[3][r] *= alpha;
      float p0 = __expf(s0[r] - nm);
      float p1 = __expf(s1[r] - nm);
      ldsP[wave][r + half * 8][sub]      = (__bf16)p0;
      ldsP[wave][r + half * 8][16 + sub] = (__bf16)p1;
      psum[r] = p0 + p1;
    }
#pragma unroll
    for (int off = 1; off < 16; off <<= 1)
#pragma unroll
      for (int r = 0; r < 8; ++r) psum[r] += __shfl_xor(psum[r], off, 32);
#pragma unroll
    for (int r = 0; r < 8; ++r) rsum[r] += psum[r];

    __builtin_amdgcn_wave_barrier();   // keep DS store->load ordered (per-wave, in-order LDS)

    // P tile (16x32) as A fragment straight from LDS (two ds b128 per lane)
    union { v16bf v; v8bf hh[2]; } pu;
    pu.hh[0] = *reinterpret_cast<const v8bf*>(&ldsP[wave][sub][half * 8]);
    pu.hh[1] = *reinterpret_cast<const v8bf*>(&ldsP[wave][sub][16 + half * 8]);
    v16bf pa = pu.v;

    // O += P @ V   (vT is [D][Mpad] so B fragments are contiguous b128 loads)
#pragma unroll
    for (int f = 0; f < 4; ++f) {
      v16bf vf = load_frag(vb, f * 16, Mpad, j0, lane);
      acc[f] = wmma_bf16(pa, vf, acc[f]);
    }
  }

  // epilogue: O /= rowsum, store bf16 into [B,N,C] layout for the proj GEMM
#pragma unroll
  for (int f = 0; f < 4; ++f)
#pragma unroll
    for (int r = 0; r < 8; ++r) {
      int qrow = q0 + r + half * 8;
      if (qrow < Nc) {
        float o = acc[f][r] / rsum[r];
        obuf[((size_t)b * Nc + qrow) * Cc + h * Dc + f * 16 + sub] = (__bf16)o;
      }
    }
}

// ---------------- Output projection GEMM: [12368 x 768 x 768] -> fp32 ----------------
__global__ __launch_bounds__(256)
void proj_gemm(const __bf16* __restrict__ A, const __bf16* __restrict__ W,
               const float* __restrict__ bias, float* __restrict__ out) {
  int lane = threadIdx.x & 31, wave = threadIdx.x >> 5;
  int wr = wave >> 2, wc = wave & 3;
  int row0 = blockIdx.x * 64 + wr * 32;
  int col0 = blockIdx.y * 256 + wc * 64;
  v8f acc[2][4];
#pragma unroll
  for (int i = 0; i < 2; ++i)
#pragma unroll
    for (int j = 0; j < 4; ++j) acc[i][j] = zero8();

  for (int kc = 0; kc < Cc; kc += 32) {
    v16bf a0 = load_frag_clamp(A, row0,      ROWS, Cc, kc, lane);
    v16bf a1 = load_frag_clamp(A, row0 + 16, ROWS, Cc, kc, lane);
#pragma unroll
    for (int j = 0; j < 4; ++j) {
      v16bf bj = load_frag(W, col0 + j * 16, Cc, kc, lane);
      acc[0][j] = wmma_bf16(a0, bj, acc[0][j]);
      acc[1][j] = wmma_bf16(a1, bj, acc[1][j]);
    }
  }

  int sub = lane & 15, half = lane >> 4;
#pragma unroll
  for (int i = 0; i < 2; ++i)
#pragma unroll
    for (int j = 0; j < 4; ++j)
#pragma unroll
      for (int r = 0; r < 8; ++r) {
        int gr = row0 + i * 16 + r + half * 8;
        if (gr >= ROWS) continue;
        int c = col0 + j * 16 + sub;
        out[(size_t)gr * Cc + c] = acc[i][j][r] + bias[c];
      }
}

// ---------------- host launcher ----------------
extern "C" void kernel_launch(void* const* d_in, const int* in_sizes, int n_in,
                              void* d_out, int out_size, void* d_ws, size_t ws_size,
                              hipStream_t stream) {
  (void)in_sizes; (void)n_in; (void)out_size; (void)ws_size;
  const float* x         = (const float*)d_in[0];
  const float* mask      = (const float*)d_in[1];
  const float* rpb       = (const float*)d_in[2];
  const float* qkv_w     = (const float*)d_in[3];
  const float* q_bias    = (const float*)d_in[4];
  const float* v_bias    = (const float*)d_in[5];
  const float* proj_w    = (const float*)d_in[6];
  const float* proj_b    = (const float*)d_in[7];
  const float* text_emb  = (const float*)d_in[8];
  const float* image_emb = (const float*)d_in[9];
  const float* pp_w1     = (const float*)d_in[10];
  const float* pp_b1     = (const float*)d_in[11];
  const float* pp_w2     = (const float*)d_in[12];
  const float* pp_b2     = (const float*)d_in[13];
  float* out = (float*)d_out;

  char* base = (char*)d_ws;
  size_t off = 0;
  auto take = [&](size_t bytes) -> char* {
    char* p = base + off;
    off += (bytes + 255) & ~size_t(255);
    return p;
  };
  __bf16* xbf   = (__bf16*)take((size_t)ROWS * Cc * 2);
  __bf16* wqkv  = (__bf16*)take((size_t)3 * Cc * Cc * 2);
  __bf16* wproj = (__bf16*)take((size_t)Cc * Cc * 2);
  float*  hbuf  = (float*) take((size_t)2 * Pc * Cc * 4);
  __bf16* qbuf  = (__bf16*)take((size_t)Bc * Hc * Npad * Dc * 2);
  __bf16* kbuf  = (__bf16*)take((size_t)Bc * Hc * Mpad * Dc * 2);
  __bf16* vT    = (__bf16*)take((size_t)Bc * Hc * Dc * Mpad * 2);
  __bf16* obuf  = (__bf16*)take((size_t)ROWS * Cc * 2);

  // zero padded regions so masked/pad lanes never inject NaN (graph-capturable)
  hipMemsetAsync(qbuf, 0, (size_t)Bc * Hc * Npad * Dc * 2, stream);
  hipMemsetAsync(kbuf, 0, (size_t)Bc * Hc * Mpad * Dc * 2, stream);
  hipMemsetAsync(vT,   0, (size_t)Bc * Hc * Dc * Mpad * 2, stream);

  { int n = ROWS * Cc;    cvt_bf16<<<(n + 255) / 256, 256, 0, stream>>>(x, xbf, n); }
  { int n = 3 * Cc * Cc;  cvt_bf16<<<(n + 255) / 256, 256, 0, stream>>>(qkv_w, wqkv, n); }
  { int n = Cc * Cc;      cvt_bf16<<<(n + 255) / 256, 256, 0, stream>>>(proj_w, wproj, n); }

  { int n = 2 * Pc * Cc;
    prompt_h<<<(n + 255) / 256, 256, 0, stream>>>(text_emb, image_emb, pp_w1, pp_b1, hbuf); }
  { int n = 2 * Pc * 2 * Cc;
    prompt_p2<<<(n + 255) / 256, 256, 0, stream>>>(hbuf, pp_w2, pp_b2, kbuf, vT); }

  { dim3 g((ROWS + 63) / 64, (3 * Cc) / 256);
    qkv_gemm<<<g, 256, 0, stream>>>(xbf, wqkv, q_bias, v_bias, qbuf, kbuf, vT); }
  { dim3 g(Bc * Hc, Npad / 64);
    attn_kernel<<<g, 128, 0, stream>>>(qbuf, kbuf, vT, rpb, mask, obuf); }
  { dim3 g((ROWS + 63) / 64, Cc / 256);
    proj_gemm<<<g, 256, 0, stream>>>(obuf, wproj, proj_b, out); }
}
